// BaselineGNN_3229815407285
// MI455X (gfx1250) — matrix operand order repaired and checked by reading, not compiled, not run on hardware
//
#include <hip/hip_runtime.h>

// ---------------------------------------------------------------------------
// BaselineGNN forward on MI455X (gfx1250), bf16 WMMA GEMM path with
// async global->LDS tile staging (ASYNCcnt) and double-buffered LDS.
//   h  = x @ We^T + be                      (GEMM, no relu)
//   for (W,b) in (W1,b1),(W2,b2):
//       aggr = segment_sum(h[col], row)    (CSR gather-accumulate)
//       h    = relu(aggr @ W^T + b)        (GEMM, relu)
//   hm = mean(h, axis=1); z = relu(hm@Wc1^T+bc1); out = z@Wc2^T+bc2
// ---------------------------------------------------------------------------

typedef __bf16 bf16;
typedef __attribute__((ext_vector_type(16))) __bf16 v16bf;
typedef __attribute__((ext_vector_type(8)))  float  v8f;

#define N_NODES 4096
#define IN_DIM  512
#define HID     4096
#define E_EDGES 65536

#define TILE_M 128
#define TILE_N 128
#define KSTEP  32
#define LDS_PAD 8

// ---- CDNA5 async global->LDS support (guarded; falls back to sync path) ----
#if defined(__HIP_DEVICE_COMPILE__) && __has_builtin(__builtin_amdgcn_global_load_async_to_lds_b128)
#define HAS_ASYNC_LDS 1
typedef __attribute__((__vector_size__(16))) int i32x4;
typedef __attribute__((address_space(1))) i32x4 as1_i32x4;
typedef __attribute__((address_space(3))) i32x4 as3_i32x4;
__device__ __forceinline__ void async_ld16(const bf16* g, bf16* l) {
    __builtin_amdgcn_global_load_async_to_lds_b128(
        (as1_i32x4*)g, (as3_i32x4*)l, 0, 0);
}
#if __has_builtin(__builtin_amdgcn_s_wait_asynccnt)
#define WAIT_ASYNC(n) __builtin_amdgcn_s_wait_asynccnt(n)
#else
#define WAIT_ASYNC(n) asm volatile("s_wait_asynccnt %0" ::"n"(n) : "memory")
#endif
#else
#define HAS_ASYNC_LDS 0
#endif

// ---------------------------------------------------------------- utilities
__global__ void zero_i32(int* __restrict__ p, int n) {
    int i = blockIdx.x * blockDim.x + threadIdx.x;
    if (i < n) p[i] = 0;
}

__global__ void cvt_f32_bf16(const float* __restrict__ in, bf16* __restrict__ out, int n4) {
    int i = blockIdx.x * blockDim.x + threadIdx.x;
    if (i < n4) {
        float4 v = *(const float4*)(in + (size_t)i * 4);
        bf16* o = out + (size_t)i * 4;
        o[0] = (bf16)v.x; o[1] = (bf16)v.y; o[2] = (bf16)v.z; o[3] = (bf16)v.w;
    }
}

// ---------------------------------------------------------------- CSR build
__global__ void count_deg(const int* __restrict__ row, int* __restrict__ deg) {
    int e = blockIdx.x * blockDim.x + threadIdx.x;
    if (e < E_EDGES) atomicAdd(&deg[row[e]], 1);
}

// one block of 1024 threads scans 4096 degrees -> exclusive row_ptr[4097]
__global__ __launch_bounds__(1024) void scan_deg(const int* __restrict__ deg,
                                                 int* __restrict__ row_ptr) {
    __shared__ int part[1024];
    int t = threadIdx.x;
    int b = t * 4;
    int s0 = deg[b], s1 = deg[b + 1], s2 = deg[b + 2], s3 = deg[b + 3];
    int sum = s0 + s1 + s2 + s3;
    part[t] = sum;
    __syncthreads();
    for (int off = 1; off < 1024; off <<= 1) {
        int v = (t >= off) ? part[t - off] : 0;
        __syncthreads();
        part[t] += v;
        __syncthreads();
    }
    int excl = part[t] - sum;
    row_ptr[b]     = excl;
    row_ptr[b + 1] = excl + s0;
    row_ptr[b + 2] = excl + s0 + s1;
    row_ptr[b + 3] = excl + s0 + s1 + s2;
    if (t == 1023) row_ptr[4096] = excl + sum;
}

__global__ void fill_csr(const int* __restrict__ row, const int* __restrict__ col,
                         const int* __restrict__ row_ptr, int* __restrict__ cursor,
                         int* __restrict__ csr_col) {
    int e = blockIdx.x * blockDim.x + threadIdx.x;
    if (e < E_EDGES) {
        int r = row[e];
        int p = atomicAdd(&cursor[r], 1);
        csr_col[row_ptr[r] + p] = col[e];
    }
}

// ------------------------------------------------------- neighbor aggregation
// one block per destination node; 512 threads x 8 contiguous features each.
// Streams ~0.5 GiB in bf16 -> pure HBM-roofline phase; prefetch next row.
__global__ __launch_bounds__(512) void aggregate(const bf16* __restrict__ h,
                                                 const int* __restrict__ row_ptr,
                                                 const int* __restrict__ csr_col,
                                                 bf16* __restrict__ aggr) {
    int node = blockIdx.x;
    int f0 = threadIdx.x * 8;              // 512*8 == 4096
    float acc[8];
#pragma unroll
    for (int j = 0; j < 8; j++) acc[j] = 0.0f;

    int s = row_ptr[node], e = row_ptr[node + 1];
    for (int i = s; i < e; i++) {
        int nb = csr_col[i];
        if (i + 1 < e) {
            int nb2 = csr_col[i + 1];
            __builtin_prefetch(h + (size_t)nb2 * HID + f0, 0, 1);
        }
        union { uint4 u; bf16 b[8]; } v;
        v.u = *(const uint4*)(h + (size_t)nb * HID + f0);
#pragma unroll
        for (int j = 0; j < 8; j++) acc[j] += (float)v.b[j];
    }
    union { uint4 u; bf16 b[8]; } o;
#pragma unroll
    for (int j = 0; j < 8; j++) o.b[j] = (bf16)acc[j];
    *(uint4*)(aggr + (size_t)node * HID + f0) = o.u;
}

// ------------------------------------------------------------- WMMA GEMM
// C[M,Nn] = act(A[M,K](bf16) @ B[Nn,K]^T(bf16) + bias), C in bf16.
// block 256 threads = 8 waves; block tile 128x128; wave tile 64x32.
// Tiles staged via async global->LDS copies, double-buffered; the steady
// state is branchless: we always issue 4 async loads for the "next" tile
// (the final iteration re-stages tile 0 into the dead buffer, never read)
// and always wait asynccnt<=4 so exactly the current tile has retired.
__global__ __launch_bounds__(256) void gemm_bf16_wmma(
    const bf16*  __restrict__ A, const bf16* __restrict__ B,
    const float* __restrict__ bias, bf16* __restrict__ C,
    int M, int Nn, int K, int do_relu) {

    __shared__ bf16 As[2][TILE_M][KSTEP + LDS_PAD];
    __shared__ bf16 Bs[2][TILE_N][KSTEP + LDS_PAD];

    const int t = threadIdx.x;
    const int lane = t & 31;
    const int wave = t >> 5;          // 0..7
    const int waveM = wave & 1;       // 2 waves along M
    const int waveN = wave >> 1;      // 4 waves along N
    const int m0 = blockIdx.y * TILE_M;
    const int n0 = blockIdx.x * TILE_N;

    const int l15 = lane & 15;
    const int hi  = (lane >= 16);
    const int akb = hi ? 16 : 0;      // A frag K base (K = akb..akb+15 contiguous, matches B)
    const int bkb = hi ? 16 : 0;      // B frag K base (K = bkb..bkb+15 contiguous, matches A)

    // per-thread staging coordinates: 512 chunks of 16B per tile / 256 threads
    const int r0 = t >> 2;            // chunk rows for i=0
    const int r1 = (t + 256) >> 2;    // chunk rows for i=1
    const int kkc = (t & 3) * 8;      // chunk K offset

    v8f acc[4][2];
#pragma unroll
    for (int i = 0; i < 4; i++)
#pragma unroll
        for (int j = 0; j < 2; j++) acc[i][j] = {};

    const int nk = K / KSTEP;

#if HAS_ASYNC_LDS
    // prologue: stage tile 0 into buffer 0 (4 async instrs per thread)
    {
        async_ld16(A + (size_t)(m0 + r0) * K + kkc, &As[0][r0][kkc]);
        async_ld16(A + (size_t)(m0 + r1) * K + kkc, &As[0][r1][kkc]);
        async_ld16(B + (size_t)(n0 + r0) * K + kkc, &Bs[0][r0][kkc]);
        async_ld16(B + (size_t)(n0 + r1) * K + kkc, &Bs[0][r1][kkc]);
    }
#endif

    for (int it = 0; it < nk; it++) {
        const int cur = it & 1;
        __syncthreads();   // all waves done computing from buffer cur^1
#if HAS_ASYNC_LDS
        {
            // branchless: last iteration re-stages k=0 into the dead buffer
            const int kn = (it + 1 < nk) ? (it + 1) * KSTEP : 0;
            const int nxt = cur ^ 1;
            async_ld16(A + (size_t)(m0 + r0) * K + kn + kkc, &As[nxt][r0][kkc]);
            async_ld16(A + (size_t)(m0 + r1) * K + kn + kkc, &As[nxt][r1][kkc]);
            async_ld16(B + (size_t)(n0 + r0) * K + kn + kkc, &Bs[nxt][r0][kkc]);
            async_ld16(B + (size_t)(n0 + r1) * K + kn + kkc, &Bs[nxt][r1][kkc]);
            WAIT_ASYNC(4);   // tile `cur` complete (async loads retire in order)
        }
#else
        // sync fallback: load tiles through VGPRs into buffer cur
        {
            const int k0 = it * KSTEP;
            *(uint4*)&As[cur][r0][kkc] = *(const uint4*)(A + (size_t)(m0 + r0) * K + k0 + kkc);
            *(uint4*)&As[cur][r1][kkc] = *(const uint4*)(A + (size_t)(m0 + r1) * K + k0 + kkc);
            *(uint4*)&Bs[cur][r0][kkc] = *(const uint4*)(B + (size_t)(n0 + r0) * K + k0 + kkc);
            *(uint4*)&Bs[cur][r1][kkc] = *(const uint4*)(B + (size_t)(n0 + r1) * K + k0 + kkc);
        }
#endif
        __syncthreads();   // tile `cur` visible to all waves

        // B fragments (lane holds column n = l15; K = bkb..bkb+15 contiguous)
        v16bf bfrag[2];
#pragma unroll
        for (int nt = 0; nt < 2; nt++) {
            int n = waveN * 32 + nt * 16 + l15;
#pragma unroll
            for (int i = 0; i < 16; i++) bfrag[nt][i] = Bs[cur][n][bkb + i];
        }
#pragma unroll
        for (int mt = 0; mt < 4; mt++) {
            int r = waveM * 64 + mt * 16 + l15;
            v16bf afrag;
#pragma unroll
            for (int i = 0; i < 8; i++) afrag[i]     = As[cur][r][akb + i];
#pragma unroll
            for (int i = 0; i < 8; i++) afrag[8 + i] = As[cur][r][akb + 8 + i];
#pragma unroll
            for (int nt = 0; nt < 2; nt++) {
                acc[mt][nt] = __builtin_amdgcn_wmma_f32_16x16x32_bf16(
                    false, afrag, false, bfrag[nt], (short)0, acc[mt][nt],
                    false, false);
            }
        }
    }
    // note: any still-inflight async loads drain at S_ENDPGM (implicit wait-idle)

    // store: 16x16 f32 C/D layout -> VGPR d: lanes 0-15 M=d, lanes 16-31 M=d+8
#pragma unroll
    for (int mt = 0; mt < 4; mt++) {
#pragma unroll
        for (int nt = 0; nt < 2; nt++) {
            int col = n0 + waveN * 32 + nt * 16 + l15;
            float bv = bias[col];
            int rbase = m0 + waveM * 64 + mt * 16 + (hi ? 8 : 0);
#pragma unroll
            for (int d = 0; d < 8; d++) {
                float v = acc[mt][nt][d] + bv;
                if (do_relu) v = v > 0.0f ? v : 0.0f;
                C[(size_t)(rbase + d) * Nn + col] = (bf16)v;
            }
        }
    }
}

// ------------------------------------------------------------- epilogue
__global__ __launch_bounds__(256) void row_mean(const bf16* __restrict__ h,
                                                float* __restrict__ hm) {
    __shared__ float red[256];
    int r = blockIdx.x, t = threadIdx.x;
    const bf16* p = h + (size_t)r * HID + t * 16;
    float s = 0.0f;
#pragma unroll
    for (int j = 0; j < 16; j++) s += (float)p[j];
    red[t] = s;
    __syncthreads();
    for (int o = 128; o > 0; o >>= 1) {
        if (t < o) red[t] += red[t + o];
        __syncthreads();
    }
    if (t == 0) hm[r] = red[0] * (1.0f / (float)HID);
}

__global__ __launch_bounds__(128) void gemv_relu(const float* __restrict__ hm,
                                                 const float* __restrict__ Wc1,
                                                 const float* __restrict__ bc1,
                                                 float* __restrict__ z) {
    __shared__ float red[128];
    int j = blockIdx.x, t = threadIdx.x;
    const float* w = Wc1 + (size_t)j * HID;
    float s = 0.0f;
    for (int k = t; k < HID; k += 128) s += hm[k] * w[k];
    red[t] = s;
    __syncthreads();
    for (int o = 64; o > 0; o >>= 1) {
        if (t < o) red[t] += red[t + o];
        __syncthreads();
    }
    if (t == 0) {
        float v = red[0] + bc1[j];
        z[j] = v > 0.0f ? v : 0.0f;
    }
}

__global__ __launch_bounds__(256) void final_dot(const float* __restrict__ z,
                                                 const float* __restrict__ Wc2,
                                                 const float* __restrict__ bc2,
                                                 float* __restrict__ out) {
    __shared__ float red[256];
    int t = threadIdx.x;
    float s = 0.0f;
    for (int k = t; k < HID / 2; k += 256) s += z[k] * Wc2[k];
    red[t] = s;
    __syncthreads();
    for (int o = 128; o > 0; o >>= 1) {
        if (t < o) red[t] += red[t + o];
        __syncthreads();
    }
    if (t == 0) out[0] = red[0] + bc2[0];
}

// ---------------------------------------------------------------------------
extern "C" void kernel_launch(void* const* d_in, const int* in_sizes, int n_in,
                              void* d_out, int out_size, void* d_ws, size_t ws_size,
                              hipStream_t stream) {
    const float* x   = (const float*)d_in[0];
    const int*   ei  = (const int*)d_in[1];     // [2, E]: row then col
    const float* We  = (const float*)d_in[2];
    const float* be  = (const float*)d_in[3];
    const float* W1  = (const float*)d_in[4];
    const float* b1  = (const float*)d_in[5];
    const float* W2  = (const float*)d_in[6];
    const float* b2  = (const float*)d_in[7];
    const float* Wc1 = (const float*)d_in[8];
    const float* bc1 = (const float*)d_in[9];
    const float* Wc2 = (const float*)d_in[10];
    const float* bc2 = (const float*)d_in[11];
    float* out = (float*)d_out;

    char* ws = (char*)d_ws;
    size_t off = 0;
    auto carve = [&](size_t bytes) {
        void* p = ws + off;
        off += (bytes + 255) & ~(size_t)255;
        return p;
    };
    bf16* x_bf   = (bf16*)carve((size_t)N_NODES * IN_DIM * sizeof(bf16)); //  4 MB
    bf16* wbuf   = (bf16*)carve((size_t)HID * HID * sizeof(bf16));        // 32 MB (We/W1/W2 reuse)
    bf16* h      = (bf16*)carve((size_t)N_NODES * HID * sizeof(bf16));    // 32 MB
    bf16* aggr   = (bf16*)carve((size_t)N_NODES * HID * sizeof(bf16));    // 32 MB
    int*  deg    = (int*)carve(N_NODES * sizeof(int));
    int*  cursor = (int*)carve(N_NODES * sizeof(int));
    int*  rowptr = (int*)carve((N_NODES + 1) * sizeof(int));
    int*  csrcol = (int*)carve(E_EDGES * sizeof(int));
    float* hm    = (float*)carve(N_NODES * sizeof(float));
    float* z     = (float*)carve((HID / 2) * sizeof(float));
    (void)ws_size; (void)in_sizes; (void)n_in; (void)out_size;

    const int* e_row = ei;
    const int* e_col = ei + E_EDGES;

    // CSR build
    zero_i32<<<(N_NODES + 255) / 256, 256, 0, stream>>>(deg, N_NODES);
    zero_i32<<<(N_NODES + 255) / 256, 256, 0, stream>>>(cursor, N_NODES);
    count_deg<<<E_EDGES / 256, 256, 0, stream>>>(e_row, deg);
    scan_deg<<<1, 1024, 0, stream>>>(deg, rowptr);
    fill_csr<<<E_EDGES / 256, 256, 0, stream>>>(e_row, e_col, rowptr, cursor, csrcol);

    dim3 ggrid(HID / TILE_N, N_NODES / TILE_M);

    // embed: h = x @ We^T + be
    cvt_f32_bf16<<<(N_NODES * IN_DIM / 4 + 255) / 256, 256, 0, stream>>>(
        x, x_bf, N_NODES * IN_DIM / 4);
    cvt_f32_bf16<<<(HID * IN_DIM / 4 + 255) / 256, 256, 0, stream>>>(
        We, wbuf, HID * IN_DIM / 4);
    gemm_bf16_wmma<<<ggrid, 256, 0, stream>>>(x_bf, wbuf, be, h,
                                              N_NODES, HID, IN_DIM, 0);

    // layer 1
    aggregate<<<N_NODES, 512, 0, stream>>>(h, rowptr, csrcol, aggr);
    cvt_f32_bf16<<<(HID * HID / 4 + 255) / 256, 256, 0, stream>>>(
        W1, wbuf, HID * HID / 4);
    gemm_bf16_wmma<<<ggrid, 256, 0, stream>>>(aggr, wbuf, b1, h,
                                              N_NODES, HID, HID, 1);
    // layer 2
    aggregate<<<N_NODES, 512, 0, stream>>>(h, rowptr, csrcol, aggr);
    cvt_f32_bf16<<<(HID * HID / 4 + 255) / 256, 256, 0, stream>>>(
        W2, wbuf, HID * HID / 4);
    gemm_bf16_wmma<<<ggrid, 256, 0, stream>>>(aggr, wbuf, b2, h,
                                              N_NODES, HID, HID, 1);

    // classifier head
    row_mean<<<N_NODES, 256, 0, stream>>>(h, hm);
    gemv_relu<<<HID / 2, 128, 0, stream>>>(hm, Wc1, bc1, z);
    final_dot<<<1, 256, 0, stream>>>(z, Wc2, bc2, out);
}
